// MLP_81003083203269
// MI455X (gfx1250) — matrix operand order repaired and checked by reading, not compiled, not run on hardware
//
#include <hip/hip_runtime.h>
#include <hip/hip_bf16.h>

typedef __attribute__((ext_vector_type(4)))  int      v4i;
typedef __attribute__((ext_vector_type(8)))  _Float16 v8h;
typedef __attribute__((ext_vector_type(16))) _Float16 v16h;
typedef __attribute__((ext_vector_type(8)))  float    v8f;

#define BM 256      // block tile M
#define BN 128      // block tile N
#define BK 64       // block tile K (two 32-deep WMMA substeps)
#define LDA 72      // LDS row stride in halves (64 payload + 8 pad = 144B, 16B-aligned)

#if __has_builtin(__builtin_amdgcn_global_load_async_to_lds_b128)
#define ASYNC_LDS 1
#else
#define ASYNC_LDS 0
#endif

#define AS1 __attribute__((address_space(1)))
#define AS3 __attribute__((address_space(3)))

// ---- 16B chunk stagers (8 halves per chunk) ------------------------------

// fp32 activations -> f16 in LDS (layer 1)
__device__ __forceinline__ void stage_chunk_A(const float* __restrict__ gp,
                                              _Float16* lp) {
    const float4* q = (const float4*)gp;
    float4 a = q[0], b = q[1];
    v8h t;
    t[0]=(_Float16)a.x; t[1]=(_Float16)a.y; t[2]=(_Float16)a.z; t[3]=(_Float16)a.w;
    t[4]=(_Float16)b.x; t[5]=(_Float16)b.y; t[6]=(_Float16)b.z; t[7]=(_Float16)b.w;
    *(v8h*)lp = t;
}

// f16 activations -> LDS: async DMA copy (ASYNCcnt path) when available
__device__ __forceinline__ void stage_chunk_A(const _Float16* __restrict__ gp,
                                              _Float16* lp) {
#if ASYNC_LDS
    __builtin_amdgcn_global_load_async_to_lds_b128(
        (AS1 v4i*)gp, (AS3 v4i*)lp, 0, 0);
#else
    *(v8h*)lp = *(const v8h*)gp;
#endif
}

__device__ __forceinline__ void wait_async_stage() {
#if ASYNC_LDS
#if __has_builtin(__builtin_amdgcn_s_wait_asynccnt)
    __builtin_amdgcn_s_wait_asynccnt(0);
#else
    asm volatile("s_wait_asynccnt 0" ::: "memory");
#endif
#endif
}

// ---- quantized GEMM: C[M,N] = act( scale[n] * (A[M,K] @ (W[N,K]-128)^T) + bias[n] )
// 256 threads = 8 waves arranged 4(M) x 2(N); wave tile 64x64 = 4x4 WMMA tiles.
// Per BK=64 stage: 2 substeps x 16 = 32 v_wmma_f32_16x16x32_f16 per wave.
template <typename TA, typename TO, bool RELU, bool HASBIAS>
__global__ __launch_bounds__(256) void qgemm_wmma(
    const TA* __restrict__ A, const int* __restrict__ Wq,
    const float* __restrict__ scale, const float* __restrict__ bias,
    TO* __restrict__ C, int M, int N, int K)
{
    __shared__ __align__(64) _Float16 As[BM * LDA];
    __shared__ __align__(64) _Float16 Ws[BN * LDA];

    const int tid  = threadIdx.x;
    const int wave = tid >> 5;
    const int lane = tid & 31;
    const int lr   = lane & 15;   // row/col within a 16x16 tile
    const int lh   = lane >> 4;   // half-wave selector

    const int wm = wave >> 1;     // 0..3 -> M offset wm*64
    const int wn = wave & 1;      // 0..1 -> N offset wn*64

    const int m0 = blockIdx.y * BM;
    const int n0 = blockIdx.x * BN;

    v8f acc[4][4] = {};

    const int ksteps = K / BK;
    for (int kt = 0; kt < ksteps; ++kt) {
        const int k0 = kt * BK;

        // ---- stage A tile: BM x BK halves, 16B chunks, coalesced (8 chunks/row)
        #pragma unroll
        for (int it = 0; it < (BM * BK / 8) / 256; ++it) {   // 8 iters
            const int idx = it * 256 + tid;
            const int row = idx >> 3;
            const int c8  = idx & 7;
            stage_chunk_A(A + (size_t)(m0 + row) * K + k0 + c8 * 8,
                          &As[row * LDA + c8 * 8]);
        }
        // ---- stage W tile: BN x BK, int32 -> f16(w-128), guard N
        #pragma unroll
        for (int it = 0; it < (BN * BK / 8) / 256; ++it) {   // 4 iters
            const int idx = it * 256 + tid;
            const int row = idx >> 3;
            const int c8  = idx & 7;
            const int n   = n0 + row;
            v8h t;
            if (n < N) {
                const int4* q = (const int4*)(Wq + (size_t)n * K + k0 + c8 * 8);
                int4 a = q[0], b = q[1];
                t[0]=(_Float16)(float)(a.x-128); t[1]=(_Float16)(float)(a.y-128);
                t[2]=(_Float16)(float)(a.z-128); t[3]=(_Float16)(float)(a.w-128);
                t[4]=(_Float16)(float)(b.x-128); t[5]=(_Float16)(float)(b.y-128);
                t[6]=(_Float16)(float)(b.z-128); t[7]=(_Float16)(float)(b.w-128);
                if (kt + 1 < ksteps)
                    __builtin_prefetch((const void*)(q + 2 * (BK / 8)), 0, 1);
            } else {
                t = (v8h)0;
            }
            *(v8h*)&Ws[row * LDA + c8 * 8] = t;
        }
        wait_async_stage();
        __syncthreads();

        // ---- two 32-deep WMMA substeps over the staged BK=64 ----
        #pragma unroll
        for (int s = 0; s < 2; ++s) {
            const int kb = s * 32;
            // A 16x32 frag: lanes 0-15 hold K[0:8]+K[16:24]; 16-31 hold K[8:16]+K[24:32]
            v16h afrag[4];
            #pragma unroll
            for (int i = 0; i < 4; ++i) {
                const _Float16* r = &As[(wm * 64 + i * 16 + lr) * LDA + kb];
                v8h lo = *(const v8h*)(r + lh * 8);
                v8h hi = *(const v8h*)(r + 16 + lh * 8);
                afrag[i] = __builtin_shufflevector(lo, hi,
                    0,1,2,3,4,5,6,7,8,9,10,11,12,13,14,15);
            }
            // B 32x16 frag: lane = column; lanes 0-15 hold K[0:8]+K[16:24]; 16-31 hold K[8:16]+K[24:32]
            v16h bfrag[4];
            #pragma unroll
            for (int j = 0; j < 4; ++j) {
                const _Float16* r = &Ws[(wn * 64 + j * 16 + lr) * LDA + kb];
                v8h lo = *(const v8h*)(r + lh * 8);
                v8h hi = *(const v8h*)(r + 16 + lh * 8);
                bfrag[j] = __builtin_shufflevector(lo, hi,
                    0,1,2,3,4,5,6,7,8,9,10,11,12,13,14,15);
            }
            #pragma unroll
            for (int i = 0; i < 4; ++i)
                #pragma unroll
                for (int j = 0; j < 4; ++j)
                    acc[i][j] = __builtin_amdgcn_wmma_f32_16x16x32_f16(
                        false, afrag[i], false, bfrag[j],
                        (short)0, acc[i][j], false, false);
        }
        __syncthreads();
    }

    // ---- epilogue: scale/bias/relu, scatter per C/D VGPR layout ----
    #pragma unroll
    for (int j = 0; j < 4; ++j) {
        const int col = n0 + wn * 64 + j * 16 + lr;
        float s = 0.f, bb = 0.f;
        if (col < N) {
            s = scale[col];
            if (HASBIAS) bb = bias[col];
        }
        #pragma unroll
        for (int i = 0; i < 4; ++i) {
            #pragma unroll
            for (int r = 0; r < 8; ++r) {
                // VGPR r: lanes 0-15 -> M=r ; lanes 16-31 -> M=8+r ; N = lane&15
                const int row = m0 + wm * 64 + i * 16 + lh * 8 + r;
                float y = s * acc[i][j][r] + bb;
                if (RELU) y = fmaxf(y, 0.f);
                if (col < N) C[(size_t)row * N + col] = (TO)y;
            }
        }
    }
}

// ---- in-place row-wise log-softmax: one block per row --------------------
__global__ __launch_bounds__(256) void logsoftmax_inplace(float* __restrict__ X, int N)
{
    const int row = blockIdx.x;
    float* x = X + (size_t)row * N;
    __shared__ float red[256];
    const int tid = threadIdx.x;

    float m = -3.402823466e+38f;
    for (int i = tid; i < N; i += 256) m = fmaxf(m, x[i]);
    red[tid] = m;
    __syncthreads();
    #pragma unroll
    for (int s = 128; s > 0; s >>= 1) {
        if (tid < s) red[tid] = fmaxf(red[tid], red[tid + s]);
        __syncthreads();
    }
    m = red[0];
    __syncthreads();

    float sum = 0.f;
    for (int i = tid; i < N; i += 256) sum += __expf(x[i] - m);
    red[tid] = sum;
    __syncthreads();
    #pragma unroll
    for (int s = 128; s > 0; s >>= 1) {
        if (tid < s) red[tid] += red[tid + s];
        __syncthreads();
    }
    const float lse = m + __logf(red[0]);

    for (int i = tid; i < N; i += 256) x[i] = x[i] - lse;
}

// -------------------------------------------------------------------------
extern "C" void kernel_launch(void* const* d_in, const int* in_sizes, int n_in,
                              void* d_out, int out_size, void* d_ws, size_t ws_size,
                              hipStream_t stream)
{
    const float* X  = (const float*)d_in[0];
    const int*   w1 = (const int*)d_in[1];
    const float* s1 = (const float*)d_in[2];
    const float* b1 = (const float*)d_in[3];
    const int*   w2 = (const int*)d_in[4];
    const float* s2 = (const float*)d_in[5];
    const float* b2 = (const float*)d_in[6];
    const int*   w3 = (const int*)d_in[7];
    const float* s3 = (const float*)d_in[8];
    const float* b3 = (const float*)d_in[9];
    const int*   w4 = (const int*)d_in[10];
    const float* s4 = (const float*)d_in[11];
    const float* b4 = (const float*)d_in[12];
    const int*   w5 = (const int*)d_in[13];
    const float* s5 = (const float*)d_in[14];

    const int B = in_sizes[0] / 1024;   // 16384
    const int D_IN = 1024, H = 4096, H4 = 1024, H8 = 512, D_OUT = 1000;

    // workspace ping-pong: region0 holds h1 (B*4096 f16) then h3 (B*1024 f16);
    // region1 holds h2 (B*1024 f16) then h4 (B*512 f16).
    _Float16* r0 = (_Float16*)d_ws;
    _Float16* r1 = (_Float16*)((char*)d_ws + (size_t)B * H * sizeof(_Float16));
    _Float16* h1 = r0;
    _Float16* h2 = r1;
    _Float16* h3 = r0;
    _Float16* h4 = r1;
    float* logits = (float*)d_out;      // fp32 logits in place, then log-softmax

    dim3 blk(256);
    auto grd = [](int n, int m) { return dim3((n + BN - 1) / BN, m / BM); };

    qgemm_wmma<float,    _Float16, true,  true ><<<grd(H,    B), blk, 0, stream>>>(X,  w1, s1, b1, h1, B, H,    D_IN);
    qgemm_wmma<_Float16, _Float16, true,  true ><<<grd(H4,   B), blk, 0, stream>>>(h1, w2, s2, b2, h2, B, H4,   H);
    qgemm_wmma<_Float16, _Float16, true,  true ><<<grd(H4,   B), blk, 0, stream>>>(h2, w3, s3, b3, h3, B, H4,   H4);
    qgemm_wmma<_Float16, _Float16, true,  true ><<<grd(H8,   B), blk, 0, stream>>>(h3, w4, s4, b4, h4, B, H8,   H4);
    qgemm_wmma<_Float16, float,    false, false><<<grd(D_OUT,B), blk, 0, stream>>>(h4, w5, s5, nullptr, logits, B, D_OUT, H8);

    logsoftmax_inplace<<<B, blk, 0, stream>>>(logits, D_OUT);
}
